// MTDecoder_7052336300230
// MI455X (gfx1250) — matrix operand order repaired and checked by reading, not compiled, not run on hardware
//
#include <hip/hip_runtime.h>

// ---------------------------------------------------------------------------
// MT decoder for MI455X (gfx1250, wave32, WMMA bf16 16x16x32).
// Key optimization: enc_x @ a1_w[512:] is hoisted out of the recurrence.
// All GEMMs run on v_wmma_f32_16x16x32_bf16 with fp32 accumulation.
// ---------------------------------------------------------------------------

typedef __attribute__((ext_vector_type(16))) __bf16 v16bf;
typedef __attribute__((ext_vector_type(8)))  __bf16 v8bf;
typedef __attribute__((ext_vector_type(8)))  float  v8f;

#define D_MODEL 512
#define T_ENC   512
#define BATCH   64
#define SEQLEN  128
#define MTOT    (T_ENC * BATCH)   // 32768 flattened attention rows

// ---------------------------------------------------------------------------
// Generic WMMA GEMM:  C[M,N] = act(A[M,K] @ B + bias),  B given pre-transposed
// as BT[N][K] bf16 (so each lane's B-fragment is a contiguous 32-byte load).
// grid = (M/16, N/128), block = 256 threads (8 waves, one 16x16 tile each).
// A is fp32 row-major; converted to bf16 into LDS cooperatively.
// ---------------------------------------------------------------------------
__global__ void gemm_wmma_bf16(const float* __restrict__ A,
                               const __bf16* __restrict__ BT,
                               const float* __restrict__ bias,
                               float* __restrict__ C,
                               int K, int kShift, int N, int relu)
{
    __shared__ __align__(32) __bf16 As[16 * 1024];  // supports K up to 1024

    const int tid  = threadIdx.x;
    const int m0   = blockIdx.x * 16;
    const int wave = tid >> 5;
    const int lane = tid & 31;
    const int half = lane >> 4;        // K-half selector (contiguous k = 16*half + s order)
    const int nloc = lane & 15;
    const int n0   = blockIdx.y * 128 + wave * 16;

    // cooperative load+convert of the 16xK A tile
    const int kMask = K - 1;
    for (int idx = tid; idx < 16 * K; idx += 256) {
        int r = idx >> kShift;
        int k = idx & kMask;
        As[idx] = (__bf16)A[(size_t)(m0 + r) * K + k];
    }
    __syncthreads();

    v8f acc = {0.f, 0.f, 0.f, 0.f, 0.f, 0.f, 0.f, 0.f};
    // A-frag: row = lane&15; k = k0 + half*16 + {0..15}
    const __bf16* arow = As + (size_t)(lane & 15) * K + half * 16;
    // B-frag: col = n0+nloc; k = k0 + half*16 + {0..15} (contiguous in BT row)
    const __bf16* brow = BT + (size_t)(n0 + nloc) * K + half * 16;

    for (int k0 = 0; k0 < K; k0 += 32) {
        v8bf alo = *(const v8bf*)(arow + k0);
        v8bf ahi = *(const v8bf*)(arow + k0 + 8);
        v16bf a = __builtin_shufflevector(alo, ahi,
                     0,1,2,3,4,5,6,7,8,9,10,11,12,13,14,15);
        v16bf b = *(const v16bf*)(brow + k0);
        acc = __builtin_amdgcn_wmma_f32_16x16x32_bf16(
                  false, a, false, b, (short)0, acc, false, false);
    }

    const float bv = bias ? bias[n0 + nloc] : 0.f;
#pragma unroll
    for (int v = 0; v < 8; ++v) {
        int m = m0 + v + half * 8;     // C/D layout: VGPR v -> row v + 8*half
        float r = acc[v] + bv;
        if (relu) r = fmaxf(r, 0.f);
        C[(size_t)m * N + n0 + nloc] = r;
    }
}

// ---------------------------------------------------------------------------
// Fused attention scores: for a 16-row tile of the flattened [T*B] dimension,
//   s1 = relu(enc_pre + y_pre[b])         (built as bf16 in LDS)
//   s2 = relu(s1 @ a2_w + a2_b)           (8 waves x WMMA, N=128)
//   logit = s2 . a3_w + a3_b
// grid = MTOT/16 = 2048 blocks, block = 256.
// ---------------------------------------------------------------------------
__global__ void attn_scores(const float* __restrict__ enc_pre,  // [MTOT][512]
                            const float* __restrict__ y_pre,    // [64][512]
                            const __bf16* __restrict__ a2T,     // [128][512]
                            const float* __restrict__ a2_b,     // [128]
                            const float* __restrict__ a3_w,     // [128]
                            const float* __restrict__ a3_b,     // [1]
                            float* __restrict__ logits)         // [MTOT]
{
    __shared__ __align__(32) __bf16 s1[16 * 512];
    __shared__ float s2[16 * 128];

    const int tid = threadIdx.x;
    const int m0  = blockIdx.x * 16;

    for (int idx = tid; idx < 16 * 512; idx += 256) {
        int r = idx >> 9, k = idx & 511;
        int m = m0 + r;
        int b = m & (BATCH - 1);            // m = t*64 + b
        float v = enc_pre[(size_t)m * 512 + k] + y_pre[b * 512 + k];
        s1[idx] = (__bf16)fmaxf(v, 0.f);
    }
    __syncthreads();

    const int wave = tid >> 5;
    const int lane = tid & 31;
    const int half = lane >> 4;
    const int nloc = lane & 15;
    const int n0   = wave * 16;

    v8f acc = {0.f, 0.f, 0.f, 0.f, 0.f, 0.f, 0.f, 0.f};
    const __bf16* arow = s1 + (size_t)(lane & 15) * 512 + half * 16;
    const __bf16* brow = a2T + (size_t)(n0 + nloc) * 512 + half * 16;
    for (int k0 = 0; k0 < 512; k0 += 32) {
        v8bf alo = *(const v8bf*)(arow + k0);
        v8bf ahi = *(const v8bf*)(arow + k0 + 8);
        v16bf a = __builtin_shufflevector(alo, ahi,
                     0,1,2,3,4,5,6,7,8,9,10,11,12,13,14,15);
        v16bf b = *(const v16bf*)(brow + k0);
        acc = __builtin_amdgcn_wmma_f32_16x16x32_bf16(
                  false, a, false, b, (short)0, acc, false, false);
    }
    const float bv = a2_b[n0 + nloc];
#pragma unroll
    for (int v = 0; v < 8; ++v) {
        int ml = v + half * 8;
        s2[ml * 128 + n0 + nloc] = fmaxf(acc[v] + bv, 0.f);
    }
    __syncthreads();

    if (tid < 16) {
        float d = a3_b[0];
        for (int j = 0; j < 128; ++j) d += s2[tid * 128 + j] * a3_w[j];
        logits[m0 + tid] = d;
    }
}

// softmax over T (axis 0) per batch column. grid = 64, block = 256.
__global__ void softmax_T(const float* __restrict__ logits, float* __restrict__ att)
{
    __shared__ float red[256];
    const int b = blockIdx.x, tid = threadIdx.x;
    float v0 = logits[tid * BATCH + b];
    float v1 = logits[(tid + 256) * BATCH + b];
    red[tid] = fmaxf(v0, v1);
    __syncthreads();
    for (int s = 128; s > 0; s >>= 1) {
        if (tid < s) red[tid] = fmaxf(red[tid], red[tid + s]);
        __syncthreads();
    }
    float mx = red[0];
    __syncthreads();
    float e0 = __expf(v0 - mx), e1 = __expf(v1 - mx);
    red[tid] = e0 + e1;
    __syncthreads();
    for (int s = 128; s > 0; s >>= 1) {
        if (tid < s) red[tid] += red[tid + s];
        __syncthreads();
    }
    float inv = 1.f / red[0];
    att[tid * BATCH + b]         = e0 * inv;
    att[(tid + 256) * BATCH + b] = e1 * inv;
}

// ctx[b] = sum_t att[t,b]*enc_x[t,b,:]; also assemble comb=[y|ctx].
// grid = (64, 2), block = 256 (thread = feature d).
__global__ void ctx_comb(const float* __restrict__ att,
                         const float* __restrict__ enc_x,
                         const float* __restrict__ y,
                         float* __restrict__ comb)
{
    const int b = blockIdx.x;
    const int d = blockIdx.y * 256 + threadIdx.x;
    float acc = 0.f;
    for (int t = 0; t < T_ENC; ++t)
        acc += att[t * BATCH + b] * enc_x[((size_t)t * BATCH + b) * D_MODEL + d];
    comb[(size_t)b * 1024 + 512 + d] = acc;
    comb[(size_t)b * 1024 + d]       = y[b * D_MODEL + d];
}

// GRU gate math: h <- (1-z)*n + z*h (in place; gh was computed from old h).
// grid = 128, block = 256 over 64*512 elements.
__global__ void gru_gates(const float* __restrict__ gi, const float* __restrict__ gh,
                          float* __restrict__ h, float* __restrict__ y_out,
                          float* __restrict__ out_t)
{
    const int idx = blockIdx.x * 256 + threadIdx.x;
    const int b = idx >> 9, d = idx & 511;
    const float* gib = gi + (size_t)b * 1536;
    const float* ghb = gh + (size_t)b * 1536;
    float r = 1.f / (1.f + __expf(-(gib[d]        + ghb[d])));
    float z = 1.f / (1.f + __expf(-(gib[512 + d]  + ghb[512 + d])));
    float n = tanhf(gib[1024 + d] + r * ghb[1024 + d]);
    float hn = (1.f - z) * n + z * h[idx];
    h[idx] = hn;
    if (y_out) y_out[idx] = hn;
    if (out_t) out_t[idx] = hn;
}

// --- prep kernels ----------------------------------------------------------
__global__ void transpose_to_bf16(const float* __restrict__ src, __bf16* __restrict__ dst,
                                  int R, int C)
{   // src [R][C] fp32 -> dst [C][R] bf16
    int idx = blockIdx.x * 256 + threadIdx.x;
    if (idx >= R * C) return;
    int r = idx / C, c = idx - r * C;
    dst[(size_t)c * R + r] = (__bf16)src[idx];
}

__global__ void convert_bf16(const float* __restrict__ src, __bf16* __restrict__ dst, int n)
{
    int idx = blockIdx.x * 256 + threadIdx.x;
    if (idx < n) dst[idx] = (__bf16)src[idx];
}

__global__ void init_state(const float* __restrict__ enc_last, float* __restrict__ y,
                           float* __restrict__ h0, float* __restrict__ h1)
{
    int idx = blockIdx.x * 256 + threadIdx.x;   // 64*512 elements
    float v = enc_last[idx];
    y[idx]  = 0.f;
    h0[idx] = v;
    h1[idx] = v;
}

// ---------------------------------------------------------------------------
extern "C" void kernel_launch(void* const* d_in, const int* in_sizes, int n_in,
                              void* d_out, int out_size, void* d_ws, size_t ws_size,
                              hipStream_t stream)
{
    const float* enc_x = (const float*)d_in[0];
    // d_in[1] = seq_len (compile-time constant 128 in this workload)
    const float* a1_w = (const float*)d_in[2];
    const float* a1_b = (const float*)d_in[3];
    const float* a2_w = (const float*)d_in[4];
    const float* a2_b = (const float*)d_in[5];
    const float* a3_w = (const float*)d_in[6];
    const float* a3_b = (const float*)d_in[7];
    const float* t1_w = (const float*)d_in[8];
    const float* t1_b = (const float*)d_in[9];
    const float* t2_w = (const float*)d_in[10];
    const float* t2_b = (const float*)d_in[11];
    const float* w_ih = (const float*)d_in[12];  // [2][1536][512]
    const float* w_hh = (const float*)d_in[13];
    const float* b_ih = (const float*)d_in[14];  // [2][1536]
    const float* b_hh = (const float*)d_in[15];
    float* out = (float*)d_out;

    char* wsb = (char*)d_ws;
    size_t off = 0;
    auto alloc = [&](size_t bytes) -> void* {
        void* p = wsb + off;
        off = (off + bytes + 255) & ~(size_t)255;
        return p;
    };

    float*  enc_pre = (float*)alloc((size_t)MTOT * 512 * 4);          // 64 MB
    __bf16* a1topT  = (__bf16*)alloc(512 * 512 * 2);
    __bf16* a1botT  = (__bf16*)alloc(512 * 512 * 2);
    __bf16* a2T     = (__bf16*)alloc(128 * 512 * 2);
    __bf16* t1T     = (__bf16*)alloc((size_t)512 * 1024 * 2);
    __bf16* t2T     = (__bf16*)alloc(512 * 512 * 2);
    __bf16* wihB    = (__bf16*)alloc((size_t)2 * 1536 * 512 * 2);     // already [N][K]
    __bf16* whhB    = (__bf16*)alloc((size_t)2 * 1536 * 512 * 2);
    float*  y       = (float*)alloc(64 * 512 * 4);
    float*  y_pre   = (float*)alloc(64 * 512 * 4);
    float*  logits  = (float*)alloc(MTOT * 4);
    float*  att     = (float*)alloc(MTOT * 4);
    float*  comb    = (float*)alloc(64 * 1024 * 4);
    float*  hidden  = (float*)alloc(64 * 512 * 4);
    float*  x       = (float*)alloc(64 * 512 * 4);
    float*  h0      = (float*)alloc(64 * 512 * 4);
    float*  h1      = (float*)alloc(64 * 512 * 4);
    float*  gi      = (float*)alloc(64 * 1536 * 4);
    float*  gh      = (float*)alloc(64 * 1536 * 4);
    (void)ws_size; (void)in_sizes; (void)n_in; (void)out_size;

    // ---- one-time prep: weight conversion + state init --------------------
    transpose_to_bf16<<<dim3((512*512+255)/256), 256, 0, stream>>>(a1_w,            a1topT, 512, 512);
    transpose_to_bf16<<<dim3((512*512+255)/256), 256, 0, stream>>>(a1_w + 512*512,  a1botT, 512, 512);
    transpose_to_bf16<<<dim3((512*128+255)/256), 256, 0, stream>>>(a2_w,            a2T,    512, 128);
    transpose_to_bf16<<<dim3((1024*512+255)/256),256, 0, stream>>>(t1_w,            t1T,   1024, 512);
    transpose_to_bf16<<<dim3((512*512+255)/256), 256, 0, stream>>>(t2_w,            t2T,    512, 512);
    convert_bf16<<<dim3((2*1536*512+255)/256), 256, 0, stream>>>(w_ih, wihB, 2*1536*512);
    convert_bf16<<<dim3((2*1536*512+255)/256), 256, 0, stream>>>(w_hh, whhB, 2*1536*512);
    init_state<<<dim3(64*512/256), 256, 0, stream>>>(
        enc_x + (size_t)(T_ENC - 1) * BATCH * D_MODEL, y, h0, h1);

    // ---- hoisted: enc_pre = enc_x @ a1_w[512:] + a1_b  (M=32768,K=512,N=512)
    gemm_wmma_bf16<<<dim3(MTOT/16, 512/128), 256, 0, stream>>>(
        enc_x, a1botT, a1_b, enc_pre, 512, 9, 512, 0);

    // ---- 128 recurrent steps ---------------------------------------------
    for (int t = 0; t < SEQLEN; ++t) {
        // y_pre = y @ a1_w[:512]
        gemm_wmma_bf16<<<dim3(64/16, 512/128), 256, 0, stream>>>(
            y, a1topT, nullptr, y_pre, 512, 9, 512, 0);
        // fused s1/s2/logit
        attn_scores<<<dim3(MTOT/16), 256, 0, stream>>>(
            enc_pre, y_pre, a2T, a2_b, a3_w, a3_b, logits);
        softmax_T<<<dim3(BATCH), 256, 0, stream>>>(logits, att);
        ctx_comb<<<dim3(BATCH, 2), 256, 0, stream>>>(att, enc_x, y, comb);
        // transform MLP
        gemm_wmma_bf16<<<dim3(64/16, 512/128), 256, 0, stream>>>(
            comb, t1T, t1_b, hidden, 1024, 10, 512, 1);
        gemm_wmma_bf16<<<dim3(64/16, 512/128), 256, 0, stream>>>(
            hidden, t2T, t2_b, x, 512, 9, 512, 1);
        // GRU layer 0
        gemm_wmma_bf16<<<dim3(64/16, 1536/128), 256, 0, stream>>>(
            x, wihB, b_ih, gi, 512, 9, 1536, 0);
        gemm_wmma_bf16<<<dim3(64/16, 1536/128), 256, 0, stream>>>(
            h0, whhB, b_hh, gh, 512, 9, 1536, 0);
        gru_gates<<<dim3(64*512/256), 256, 0, stream>>>(gi, gh, h0, nullptr, nullptr);
        // GRU layer 1 (input = new h0); writes h1, next y, and step output
        gemm_wmma_bf16<<<dim3(64/16, 1536/128), 256, 0, stream>>>(
            h0, wihB + (size_t)1536*512, b_ih + 1536, gi, 512, 9, 1536, 0);
        gemm_wmma_bf16<<<dim3(64/16, 1536/128), 256, 0, stream>>>(
            h1, whhB + (size_t)1536*512, b_hh + 1536, gh, 512, 9, 1536, 0);
        gru_gates<<<dim3(64*512/256), 256, 0, stream>>>(
            gi, gh, h1, y, out + (size_t)t * BATCH * D_MODEL);
    }
}
